// CvxDifflayer_44753559224646
// MI455X (gfx1250) — compile-verified, matrix operand order checked by reading, not run on hardware
//
#include <hip/hip_runtime.h>
#include <hip/hip_bf16.h>

// PDHG LP solver for 12x12 grid graph, batch=512, 300 iterations.
// N=144 nodes, V=1012 directed edges, VN=1156 primal vars (padded to 1184),
// 2N=288 dual vars. K (288 x 1184, entries in {-1,0,1}) is exact in fp16.
// One workgroup = 16 batch rows, full iteration loop in LDS, matmuls via
// v_wmma_f32_16x16x32_f16. B operands pre-swizzled into register layout so
// the hot loop is pure ds_load_b128 / global_load_b128 / v_wmma.

typedef __attribute__((ext_vector_type(16))) _Float16 v16h;
typedef __attribute__((ext_vector_type(8)))  _Float16 v8h;
typedef __attribute__((ext_vector_type(8)))  float    v8f;

#define NN   144            // nodes
#define VE   1012           // directed edges
#define VN   1156           // V + N (true primal dim)
#define WP   1184           // padded primal dim: 74*16 = 37*32
#define M2   288            // 2N dual dim: 18*16 = 9*32
#define NT1  74             // GEMM1 output tiles (WP/16)
#define KC1  9              // GEMM1 K chunks  (M2/32)
#define NT2  18             // GEMM2 output tiles (M2/16)
#define KC2  37             // GEMM2 K chunks  (WP/32)
#define FRAG 512            // 32 lanes * 16 halves per B fragment
#define NFRAG1 (KC1*NT1)    // 666
#define NFRAG2 (KC2*NT2)    // 666
#define NITERS 300
#define NWAVE  8

// LDS row strides padded for bank-conflict-free ds_load_b128 / b32:
// fp16 strides must be multiples of 8 (16B alignment); DW stride mod 64
// chosen coprime-ish so 16 lanes cover distinct banks.
#define UBST 1192           // ubh fp16 stride: 2384B = 596 DW = 20 mod 64
#define YHST 296            // yhh fp16 stride:  592B = 148 DW = 20 mod 64
#define USF  1188           // uS  f32  stride: 1188 DW = 36 mod 64
#define YSF  292            // yS  f32  stride:  292 DW = 36 mod 64

#define LDS_US   0
#define LDS_YS   (LDS_US  + 16*USF*4)        //  76032
#define LDS_UBH  (LDS_YS  + 16*YSF*4)        //  94720
#define LDS_YHH  (LDS_UBH + 16*UBST*2)       // 132864
#define LDS_QS   (LDS_YHH + 16*YHST*2)       // 142336
#define LDS_HS   (LDS_QS  + 16*NN*4)         // 151552
#define LDS_TOT  (LDS_HS  + M2*4)            // 152704

// tau = sigma = 0.9 / sqrt(max_colsum(|K|) * max_rowsum(|K|)) = 0.9/sqrt(3*16)
#define TAU 0.12990381f
#define SIG 0.12990381f

union HF16 { v16h v; v8h h[2]; };

__device__ __forceinline__ float kelem(const float* __restrict__ A,
                                       const float* __restrict__ Ap,
                                       int i, int j) {
  // K = [[A, 0], [A_pos, -I]], zero-padded to 288 x 1184.
  if (j >= VN) return 0.0f;
  if (j < VE)  return (i < NN) ? A[i * VE + j] : Ap[(i - NN) * VE + j];
  int jz = j - VE;
  if (i < NN) return 0.0f;
  return ((i - NN) == jz) ? -1.0f : 0.0f;
}

// Pre-swizzle K and K^T into WMMA B-operand fragment layout:
// fragment (kc, nt), lane l, slot s  ->  B element (krow, ncol) with
//   base = (l>>4)*8 ; kidx = s<8 ? base+s : base+16+(s-8) ; krow = kc*32+kidx
//   ncol = nt*16 + (l&15)
__global__ void prep_k_frags(const float* __restrict__ A,
                             const float* __restrict__ Ap,
                             _Float16* __restrict__ Kh,
                             _Float16* __restrict__ KTh) {
  int idx = blockIdx.x * blockDim.x + threadIdx.x;
  if (idx < NFRAG1 * FRAG) {
    int f = idx / FRAG, r = idx % FRAG;
    int lane = r >> 4, s = r & 15;
    int kc = f / NT1, nt = f % NT1;
    int base = (lane >> 4) * 8;
    int kidx = (s < 8) ? (base + s) : (base + 16 + (s - 8));
    int i = kc * 32 + kidx;          // K row   [0,288)
    int j = nt * 16 + (lane & 15);   // K col   [0,1184)
    Kh[idx] = (_Float16)kelem(A, Ap, i, j);
  } else {
    int t = idx - NFRAG1 * FRAG;
    if (t < NFRAG2 * FRAG) {
      int f = t / FRAG, r = t % FRAG;
      int lane = r >> 4, s = r & 15;
      int kc = f / NT2, nt = f % NT2;
      int base = (lane >> 4) * 8;
      int kidx = (s < 8) ? (base + s) : (base + 16 + (s - 8));
      int kt_row = kc * 32 + kidx;        // K^T row = K col [0,1184)
      int kt_col = nt * 16 + (lane & 15); // K^T col = K row [0,288)
      KTh[t] = (_Float16)kelem(A, Ap, kt_col, kt_row);
    }
  }
}

__launch_bounds__(256)
__global__ void pdhg_kernel(const float* __restrict__ Wts,     // (512,144)
                            const float* __restrict__ bvec,    // (144)
                            const _Float16* __restrict__ Kh,
                            const _Float16* __restrict__ KTh,
                            float* __restrict__ out) {
  extern __shared__ char smem[];
  float*    uS  = (float*)(smem + LDS_US);     // 16 rows, stride USF f32
  float*    yS  = (float*)(smem + LDS_YS);     // 16 rows, stride YSF f32
  _Float16* ubh = (_Float16*)(smem + LDS_UBH); // 16 rows, stride UBST f16
  _Float16* yhh = (_Float16*)(smem + LDS_YHH); // 16 rows, stride YHST f16
  float*    qS  = (float*)(smem + LDS_QS);     // 16 x 144 f32 (weights tile)
  float*    hS  = (float*)(smem + LDS_HS);     // 288 f32: h = (b, zeros)

  const int tid  = threadIdx.x;
  const int wave = tid >> 5;
  const int lane = tid & 31;
  const int ln   = lane & 15;        // A row / C-D column / B column
  const int hb   = (lane >> 4) << 3; // K-base for A/B, M-base for C/D
  const int row0 = blockIdx.x * 16;  // batch tile origin

  // ---- one-time staging: state zeroing + q, h into LDS -------------------
  for (int i = tid; i < 16 * WP; i += 256) uS[(i / WP) * USF + (i % WP)] = 0.0f;
  for (int i = tid; i < 16 * M2; i += 256) {
    yS[(i / M2) * YSF + (i % M2)]  = 0.0f;
    yhh[(i / M2) * YHST + (i % M2)] = (_Float16)0.0f;
  }
  for (int i = tid; i < 16 * NN; i += 256)
    qS[i] = Wts[(row0 + i / NN) * NN + (i % NN)];
  for (int i = tid; i < M2; i += 256) hS[i] = (i < NN) ? bvec[i] : 0.0f;
  __syncthreads();

  for (int it = 0; it < NITERS; ++it) {
    // ---- GEMM1: t1 = y @ K  (16 x 1184), then u prox update -------------
    for (int nt = wave; nt < NT1; nt += NWAVE) {
      v8f acc = {};
      #pragma unroll
      for (int kc = 0; kc < KC1; ++kc) {
        HF16 a, b;
        const _Float16* ap = yhh + ln * YHST + kc * 32 + hb;
        a.h[0] = *(const v8h*)ap;
        a.h[1] = *(const v8h*)(ap + 16);
        const _Float16* bp = Kh + (((size_t)(kc * NT1 + nt)) * 32 + lane) * 16;
        b.h[0] = *(const v8h*)bp;
        b.h[1] = *(const v8h*)(bp + 8);
        acc = __builtin_amdgcn_wmma_f32_16x16x32_f16(
            false, a.v, false, b.v, (short)0, acc, false, false);
      }
      const int  col   = nt * 16 + ln;
      const bool qlive = (col >= VE) && (col < VN);
      const int  qoff  = qlive ? (col - VE) : 0;   // lane-uniform across r
      const float qm   = qlive ? 1.0f : 0.0f;
      const bool pad   = (col >= VN);
      #pragma unroll
      for (int r = 0; r < 8; ++r) {
        const int m = hb + r;
        float uo = uS[m * USF + col];
        float qv = qm * qS[m * NN + qoff];
        float un = uo - TAU * (qv + acc[r]);
        un = fminf(fmaxf(un, 0.0f), 1.0f);
        if (pad) un = 0.0f;                    // keep padding exactly zero
        uS[m * USF + col]   = un;
        ubh[m * UBST + col] = (_Float16)(2.0f * un - uo);   // u_bar
      }
    }
    __syncthreads();

    // ---- GEMM2: t2 = u_bar @ K^T (16 x 288), then y prox update ---------
    for (int nt = wave; nt < NT2; nt += NWAVE) {
      v8f acc = {};
      #pragma unroll 4
      for (int kc = 0; kc < KC2; ++kc) {
        HF16 a, b;
        const _Float16* ap = ubh + ln * UBST + kc * 32 + hb;
        a.h[0] = *(const v8h*)ap;
        a.h[1] = *(const v8h*)(ap + 16);
        const _Float16* bp = KTh + (((size_t)(kc * NT2 + nt)) * 32 + lane) * 16;
        b.h[0] = *(const v8h*)bp;
        b.h[1] = *(const v8h*)(bp + 8);
        acc = __builtin_amdgcn_wmma_f32_16x16x32_f16(
            false, a.v, false, b.v, (short)0, acc, false, false);
      }
      const int   col = nt * 16 + ln;
      const float hv  = hS[col];               // h = (b, zeros), lane-uniform
      #pragma unroll
      for (int r = 0; r < 8; ++r) {
        const int m = hb + r;
        float yn = yS[m * YSF + col] + SIG * (acc[r] - hv);
        yn = fmaxf(yn, 0.0f);
        yS[m * YSF + col]   = yn;
        yhh[m * YHST + col] = (_Float16)yn;
      }
    }
    __syncthreads();
  }

  // ---- output: z = u[:, V:V+N] reshaped (B,12,12) -----------------------
  for (int i = tid; i < 16 * NN; i += 256) {
    int m = i / NN, j = i % NN;
    out[(row0 + m) * NN + j] = uS[m * USF + VE + j];
  }
}

extern "C" void kernel_launch(void* const* d_in, const int* in_sizes, int n_in,
                              void* d_out, int out_size, void* d_ws, size_t ws_size,
                              hipStream_t stream) {
  (void)in_sizes; (void)n_in; (void)out_size; (void)ws_size;
  const float* Wts  = (const float*)d_in[0];   // weights (512,12,12)
  const float* A    = (const float*)d_in[1];   // (144,1012)
  const float* Apos = (const float*)d_in[2];   // (144,1012)
  const float* bvec = (const float*)d_in[3];   // (144)

  _Float16* Kh  = (_Float16*)d_ws;             // 666*512 halves
  _Float16* KTh = Kh + (size_t)NFRAG1 * FRAG;  // 666*512 halves

  const int prepN = (NFRAG1 + NFRAG2) * FRAG;  // 681,984 elements
  prep_k_frags<<<(prepN + 255) / 256, 256, 0, stream>>>(A, Apos, Kh, KTh);

  static int attr_set = 0;
  if (!attr_set) {
    hipFuncSetAttribute((const void*)pdhg_kernel,
                        hipFuncAttributeMaxDynamicSharedMemorySize, LDS_TOT);
    attr_set = 1;
  }
  pdhg_kernel<<<512 / 16, 256, LDS_TOT, stream>>>(Wts, bvec, Kh, KTh,
                                                  (float*)d_out);
}
